// DecoderRNN_Attn_4105988735706
// MI455X (gfx1250) — compile-verified, matrix operand order checked by reading, not run on hardware
//
#include <hip/hip_runtime.h>
#include <math.h>

// ---------------------------------------------------------------------------
// DecoderRNN with Luong attention, CDNA5 (gfx1250) wave32 + WMMA bf16.
// Dominant op: per-step [32x512] @ [512x32000] output projection, done with
// v_wmma_f32_16x16x32_bf16.  W_o converted once to bf16 (32.8MB, L2-resident
// on MI455X's 192MB L2).  Sequential 31-step chain -> latency bound; keep
// per-step kernels skinny and launch everything on `stream`.
// ---------------------------------------------------------------------------

#define Bq 32
#define Tq 32
#define Sq 64
#define Hq 512
#define Eq 512
#define Vq 32000

typedef __attribute__((ext_vector_type(16))) __bf16 v16bf;
typedef __attribute__((ext_vector_type(8)))  float  v8f;

// ---- bf16 helpers (manual bit ops; avoid relying on compiler conversions) --
__device__ __forceinline__ __bf16 f2bf(float f) {
    union { float f; unsigned u; } a; a.f = f;
    unsigned r = a.u + 0x7FFFu + ((a.u >> 16) & 1u);   // round-to-nearest-even
    union { unsigned short s; __bf16 b; } o;
    o.s = (unsigned short)(r >> 16);
    return o.b;
}
__device__ __forceinline__ float bf2f(__bf16 b) {
    union { unsigned short s; __bf16 b; } i; i.b = b;
    union { unsigned u; float f; } o; o.u = ((unsigned)i.s) << 16;
    return o.f;
}
__device__ __forceinline__ float sigmoidf_(float x) { return 1.0f / (1.0f + expf(-x)); }

// ---- WMMA tile loader ------------------------------------------------------
// Loads a 16x32 bf16 tile (A layout).  Also used for B = W^T where W is
// row-major NxK: B[k][n] = W[n][k], so lane n reads W row n contiguously.
// Per CDNA5 ISA 16-bit A 16x32 layout: lanes 0-15 hold K=0..7 (VGPR0-3) and
// K=16..23 (VGPR4-7); lanes 16-31 hold K=8..15 and K=24..31.
__device__ __forceinline__ v16bf load_tile16x32(const __bf16* __restrict__ base, int ld) {
    int lane = threadIdx.x & 31;
    const __bf16* p = base + (lane & 15) * ld + ((lane >> 4) << 3);
    union { v16bf v; uint4 q[2]; } u;
    u.q[0] = *reinterpret_cast<const uint4*>(p);        // K = half*8 + 0..7
    u.q[1] = *reinterpret_cast<const uint4*>(p + 16);   // K = 16 + half*8 + 0..7
    return u.v;
}

// ---------------------------------------------------------------------------
// Generic skinny GEMM:  out[32xN] = X[32xK] @ W[NxK]^T + bias
// One wave per 16-col tile, both 16-row halves of M=32.  K % 32 == 0.
// ---------------------------------------------------------------------------
__global__ __launch_bounds__(256) void wmma_gemm_f32out(
    const __bf16* __restrict__ X, const __bf16* __restrict__ W,
    const float* __restrict__ bias, float* __restrict__ out, int K, int N) {
    int wave  = threadIdx.x >> 5;
    int tile  = blockIdx.x * (blockDim.x >> 5) + wave;
    int n0    = tile * 16;
    if (n0 >= N) return;                       // wave-uniform: EXEC stays all-ones
    int lane = threadIdx.x & 31;
    int n = lane & 15, half = lane >> 4;
    float bv = bias ? bias[n0 + n] : 0.0f;
    v8f acc0 = {bv, bv, bv, bv, bv, bv, bv, bv};
    v8f acc1 = acc0;
    const __bf16* Wrow = W + (size_t)n0 * K;
    for (int k = 0; k < K; k += 32) {
        v16bf a0 = load_tile16x32(X + k,            K);   // rows 0-15
        v16bf a1 = load_tile16x32(X + 16 * K + k,   K);   // rows 16-31
        v16bf bt = load_tile16x32(Wrow + k,         K);   // B = W^T tile
        acc0 = __builtin_amdgcn_wmma_f32_16x16x32_bf16(false, a0, false, bt, (short)0, acc0, false, false);
        acc1 = __builtin_amdgcn_wmma_f32_16x16x32_bf16(false, a1, false, bt, (short)0, acc1, false, false);
    }
#pragma unroll
    for (int v = 0; v < 8; ++v) {
        int m = v + 8 * half;                  // D layout: VGPR v -> row v (+8 for upper lanes)
        out[(size_t)m * N + n0 + n]        = acc0[v];
        out[(size_t)(16 + m) * N + n0 + n] = acc1[v];
    }
}

// Same GEMM but applies tanh and stores bf16 (for ah2 = tanh([h2|ctx] @ W_c^T + b_c))
__global__ __launch_bounds__(128) void wmma_gemm_tanh_bf16out(
    const __bf16* __restrict__ X, const __bf16* __restrict__ W,
    const float* __restrict__ bias, __bf16* __restrict__ out, int K, int N) {
    int wave  = threadIdx.x >> 5;
    int tile  = blockIdx.x * (blockDim.x >> 5) + wave;
    int n0    = tile * 16;
    if (n0 >= N) return;
    int lane = threadIdx.x & 31;
    int n = lane & 15, half = lane >> 4;
    float bv = bias ? bias[n0 + n] : 0.0f;
    v8f acc0 = {bv, bv, bv, bv, bv, bv, bv, bv};
    v8f acc1 = acc0;
    const __bf16* Wrow = W + (size_t)n0 * K;
    for (int k = 0; k < K; k += 32) {
        v16bf a0 = load_tile16x32(X + k,          K);
        v16bf a1 = load_tile16x32(X + 16 * K + k, K);
        v16bf bt = load_tile16x32(Wrow + k,       K);
        acc0 = __builtin_amdgcn_wmma_f32_16x16x32_bf16(false, a0, false, bt, (short)0, acc0, false, false);
        acc1 = __builtin_amdgcn_wmma_f32_16x16x32_bf16(false, a1, false, bt, (short)0, acc1, false, false);
    }
#pragma unroll
    for (int v = 0; v < 8; ++v) {
        int m = v + 8 * half;
        out[(size_t)m * N + n0 + n]        = f2bf(tanhf(acc0[v]));
        out[(size_t)(16 + m) * N + n0 + n] = f2bf(tanhf(acc1[v]));
    }
}

// ---------------------------------------------------------------------------
// One-time prep kernels
// ---------------------------------------------------------------------------
__global__ void prep_wcat_kernel(const float* __restrict__ W_ih, const float* __restrict__ W_hh,
                                 __bf16* __restrict__ Wcat) {
    int idx = blockIdx.x * blockDim.x + threadIdx.x;   // 2048*1536
    if (idx >= 4 * Hq * (Eq + 2 * Hq)) return;
    int n = idx / 1536, k = idx % 1536;
    float v = (k < 1024) ? W_ih[(size_t)n * 1024 + k] : W_hh[(size_t)n * 512 + (k - 1024)];
    Wcat[idx] = f2bf(v);
}

__global__ void f32_to_bf16_kernel(const float* __restrict__ src, __bf16* __restrict__ dst, size_t n) {
    size_t i = (size_t)blockIdx.x * blockDim.x + threadIdx.x;
    if (i < n) dst[i] = f2bf(src[i]);
}

__global__ void bias_comb_kernel(const float* __restrict__ b_ih, const float* __restrict__ b_hh,
                                 float* __restrict__ bcomb) {
    int i = blockIdx.x * blockDim.x + threadIdx.x;
    if (i < 4 * Hq) bcomb[i] = b_ih[i] + b_hh[i];
}

// proj[b][s][k] = sum_h W_a[k][h] * enc[s][b][h]   (one-time, 0.5 GFLOP, VALU)
__global__ void proj_kernel(const float* __restrict__ W_a, const float* __restrict__ enc,
                            float* __restrict__ proj) {
    int idx = blockIdx.x * blockDim.x + threadIdx.x;   // 32*64*512
    if (idx >= Bq * Sq * Hq) return;
    int k = idx & 511, s = (idx >> 9) & 63, b = idx >> 15;
    const float* wa = W_a + (size_t)k * Hq;
    const float* eb = enc + ((size_t)s * Bq + b) * Hq;
    float acc = 0.0f;
    for (int h = 0; h < Hq; ++h) acc += wa[h] * eb[h];
    proj[((size_t)b * Sq + s) * Hq + k] = acc;
}

// ---------------------------------------------------------------------------
// Per-step kernels
// ---------------------------------------------------------------------------
// A_in[b] = [ emb[tgt[b][t]] (512) | ah (512, bf16 or 0 at t=0) | h_prev (512) ]
__global__ void pack_kernel(const int* __restrict__ tgt, const float* __restrict__ emb,
                            const __bf16* __restrict__ ah_bf, const float* __restrict__ h_prev,
                            __bf16* __restrict__ A_in, int t, int first) {
    int idx = blockIdx.x * blockDim.x + threadIdx.x;   // 32*1536
    if (idx >= Bq * 1536) return;
    int b = idx / 1536, col = idx % 1536;
    float v;
    if (col < 512) {
        int w = tgt[b * Tq + t];
        v = emb[(size_t)w * Eq + col];
    } else if (col < 1024) {
        v = first ? 0.0f : bf2f(ah_bf[(size_t)b * Hq + (col - 512)]);
    } else {
        v = h_prev[(size_t)b * Hq + (col - 1024)];
    }
    A_in[idx] = f2bf(v);
}

// LSTM elementwise (torch gate order i,f,g,o); writes fp32 h/c + bf16 h into hc_cat
__global__ void lstm_kernel(const float* __restrict__ gates, const float* __restrict__ c_prev,
                            float* __restrict__ h_out, float* __restrict__ c_out,
                            __bf16* __restrict__ hc_bf) {
    int idx = blockIdx.x * blockDim.x + threadIdx.x;   // 32*512
    if (idx >= Bq * Hq) return;
    int b = idx >> 9, n = idx & 511;
    const float* g = gates + (size_t)b * 4 * Hq;
    float iv = sigmoidf_(g[n]);
    float fv = sigmoidf_(g[512 + n]);
    float gv = tanhf(g[1024 + n]);
    float ov = sigmoidf_(g[1536 + n]);
    float c2 = fv * c_prev[idx] + iv * gv;
    float h2 = ov * tanhf(c2);
    c_out[idx] = c2;
    h_out[idx] = h2;
    hc_bf[(size_t)b * 1024 + n] = f2bf(h2);
}

// Attention per batch row: scores = h2 . proj[b], masked softmax, ctx -> hc_cat[512:]
__global__ __launch_bounds__(256) void attn_kernel(
    const float* __restrict__ h2, const float* __restrict__ proj,
    const float* __restrict__ enc, const int* __restrict__ len_src,
    __bf16* __restrict__ hc_bf) {
    int b = blockIdx.x;
    __shared__ float sh_h[Hq];
    __shared__ float sh_sc[Sq];
    int tid = threadIdx.x;
    for (int i = tid; i < Hq; i += 256) sh_h[i] = h2[(size_t)b * Hq + i];
    __syncthreads();
    if (tid < Sq) {
        const float* pr = proj + ((size_t)b * Sq + tid) * Hq;
        float acc = 0.0f;
        for (int k = 0; k < Hq; ++k) acc += sh_h[k] * pr[k];
        sh_sc[tid] = (tid < len_src[b]) ? acc : -1e9f;
    }
    __syncthreads();
    if (tid == 0) {
        float mx = sh_sc[0];
        for (int s = 1; s < Sq; ++s) mx = fmaxf(mx, sh_sc[s]);
        float sum = 0.0f;
        for (int s = 0; s < Sq; ++s) { float e = expf(sh_sc[s] - mx); sh_sc[s] = e; sum += e; }
        float inv = 1.0f / sum;
        for (int s = 0; s < Sq; ++s) sh_sc[s] *= inv;
    }
    __syncthreads();
    for (int h = tid; h < Hq; h += 256) {
        float acc = 0.0f;
        for (int s = 0; s < Sq; ++s) acc += sh_sc[s] * enc[((size_t)s * Bq + b) * Hq + h];
        hc_bf[(size_t)b * 1024 + 512 + h] = f2bf(acc);
    }
}

// log_softmax over V=32000 + argmax; one block per batch row
__global__ __launch_bounds__(256) void logsoftmax_kernel(
    const float* __restrict__ logits, float* __restrict__ out_t, float* __restrict__ words_t) {
    int b = blockIdx.x;
    const float* row = logits + (size_t)b * Vq;
    __shared__ float smax[256];
    __shared__ int   sarg[256];
    __shared__ float ssum[256];
    int tid = threadIdx.x;
    float mx = -INFINITY; int arg = 0;
    for (int v = tid; v < Vq; v += 256) {
        float x = row[v];
        if (x > mx) { mx = x; arg = v; }
    }
    smax[tid] = mx; sarg[tid] = arg;
    __syncthreads();
    for (int s = 128; s > 0; s >>= 1) {
        if (tid < s) {
            if (smax[tid + s] > smax[tid] ||
                (smax[tid + s] == smax[tid] && sarg[tid + s] < sarg[tid])) {
                smax[tid] = smax[tid + s]; sarg[tid] = sarg[tid + s];
            }
        }
        __syncthreads();
    }
    float gmax = smax[0];
    float sum = 0.0f;
    for (int v = tid; v < Vq; v += 256) sum += expf(row[v] - gmax);
    ssum[tid] = sum;
    __syncthreads();
    for (int s = 128; s > 0; s >>= 1) {
        if (tid < s) ssum[tid] += ssum[tid + s];
        __syncthreads();
    }
    float lse = gmax + logf(ssum[0]);
    for (int v = tid; v < Vq; v += 256) out_t[(size_t)b * Vq + v] = row[v] - lse;
    if (tid == 0) words_t[b] = (float)sarg[0];
}

// ---------------------------------------------------------------------------
// Host side
// ---------------------------------------------------------------------------
static inline size_t align256(size_t x) { return (x + 255) & ~(size_t)255; }

extern "C" void kernel_launch(void* const* d_in, const int* in_sizes, int n_in,
                              void* d_out, int out_size, void* d_ws, size_t ws_size,
                              hipStream_t stream) {
    const int*   tgt   = (const int*)  d_in[0];
    const int*   lens  = (const int*)  d_in[1];
    const float* enc   = (const float*)d_in[2];
    const float* h0    = (const float*)d_in[3];
    const float* c0    = (const float*)d_in[4];
    const float* emb   = (const float*)d_in[5];
    const float* W_ih  = (const float*)d_in[6];
    const float* W_hh  = (const float*)d_in[7];
    const float* b_ih  = (const float*)d_in[8];
    const float* b_hh  = (const float*)d_in[9];
    const float* W_a   = (const float*)d_in[10];
    const float* W_c   = (const float*)d_in[11];
    const float* b_c   = (const float*)d_in[12];
    const float* W_o   = (const float*)d_in[13];
    const float* b_o   = (const float*)d_in[14];
    float* out = (float*)d_out;

    // Workspace carve-out (~49 MB total)
    char* ws = (char*)d_ws;
    size_t off = 0;
    __bf16* WCAT  = (__bf16*)(ws + off); off = align256(off + (size_t)2048 * 1536 * 2);
    __bf16* WC_BF = (__bf16*)(ws + off); off = align256(off + (size_t)512 * 1024 * 2);
    __bf16* WO_BF = (__bf16*)(ws + off); off = align256(off + (size_t)Vq * Hq * 2);
    float*  BCOMB = (float*) (ws + off); off = align256(off + (size_t)2048 * 4);
    float*  PROJ  = (float*) (ws + off); off = align256(off + (size_t)Bq * Sq * Hq * 4);
    __bf16* AIN   = (__bf16*)(ws + off); off = align256(off + (size_t)Bq * 1536 * 2);
    float*  GATES = (float*) (ws + off); off = align256(off + (size_t)Bq * 2048 * 4);
    float*  Hbuf  = (float*) (ws + off); off = align256(off + (size_t)Bq * Hq * 4);
    float*  Cbuf  = (float*) (ws + off); off = align256(off + (size_t)Bq * Hq * 4);
    __bf16* HCBF  = (__bf16*)(ws + off); off = align256(off + (size_t)Bq * 1024 * 2);
    __bf16* AHBF  = (__bf16*)(ws + off); off = align256(off + (size_t)Bq * Hq * 2);
    float*  LOGT  = (float*) (ws + off); off = align256(off + (size_t)Bq * Vq * 4);
    (void)ws_size;

    // ---- one-time prep (re-run every call; deterministic) ----
    prep_wcat_kernel<<<(2048 * 1536 + 255) / 256, 256, 0, stream>>>(W_ih, W_hh, WCAT);
    f32_to_bf16_kernel<<<(int)(((size_t)512 * 1024 + 255) / 256), 256, 0, stream>>>(W_c, WC_BF, (size_t)512 * 1024);
    f32_to_bf16_kernel<<<(int)(((size_t)Vq * Hq + 255) / 256), 256, 0, stream>>>(W_o, WO_BF, (size_t)Vq * Hq);
    bias_comb_kernel<<<(2048 + 255) / 256, 256, 0, stream>>>(b_ih, b_hh, BCOMB);
    proj_kernel<<<(Bq * Sq * Hq + 255) / 256, 256, 0, stream>>>(W_a, enc, PROJ);

    float* words = out + (size_t)(Tq - 1) * Bq * Vq;

    // ---- sequential decode: 31 steps ----
    for (int t = 0; t < Tq - 1; ++t) {
        const float* h_prev = (t == 0) ? h0 : Hbuf;
        const float* c_prev = (t == 0) ? c0 : Cbuf;

        pack_kernel<<<(Bq * 1536 + 255) / 256, 256, 0, stream>>>(
            tgt, emb, AHBF, h_prev, AIN, t, t == 0 ? 1 : 0);

        // gates = A_in[32x1536] @ Wcat[2048x1536]^T + (b_ih+b_hh)
        wmma_gemm_f32out<<<16, 256, 0, stream>>>(AIN, WCAT, BCOMB, GATES, 1536, 2048);

        lstm_kernel<<<(Bq * Hq + 255) / 256, 256, 0, stream>>>(
            GATES, c_prev, Hbuf, Cbuf, HCBF);

        attn_kernel<<<Bq, 256, 0, stream>>>(Hbuf, PROJ, enc, lens, HCBF);

        // ah2 = tanh([h2|ctx][32x1024] @ W_c[512x1024]^T + b_c) -> bf16
        wmma_gemm_tanh_bf16out<<<8, 128, 0, stream>>>(HCBF, WC_BF, b_c, AHBF, 1024, 512);

        // logits = ah2[32x512] @ W_o[32000x512]^T + b_o   (dominant GEMM)
        wmma_gemm_f32out<<<250, 256, 0, stream>>>(AHBF, WO_BF, b_o, LOGT, 512, Vq);

        logsoftmax_kernel<<<Bq, 256, 0, stream>>>(
            LOGT, out + (size_t)t * Bq * Vq, words + (size_t)t * Bq);
    }
}
